// FCOSDMLNegHead2_73349451481902
// MI455X (gfx1250) — matrix operand order repaired and checked by reading, not compiled, not run on hardware
//
#include <hip/hip_runtime.h>
#include <hip/hip_bf16.h>
#include <math.h>

typedef _Float16 half_t;
typedef __attribute__((ext_vector_type(16))) _Float16 v16h;
typedef __attribute__((ext_vector_type(8)))  _Float16 v8h;
typedef __attribute__((ext_vector_type(8)))  float    v8f;

#define HH 96
#define WW 96
#define HWSZ 9216          // 96*96
#define BB 8
#define NPIX 73728         // BB*HWSZ
#define EMB1C 256
#define EMB2C 128
#define NCLSC 80
#define NCOLS 320          // NCLS * (1 pos + 3 neg)

// ---------------------------------------------------------------------------
// Weight packing: Wp[n][k], k = tap*CIN + cin  (tap = ky*3+kx), f16
// ---------------------------------------------------------------------------
__global__ __launch_bounds__(256) void pack_w_kernel(const float* __restrict__ w,
                                                     half_t* __restrict__ Wp,
                                                     int COUT, int CIN) {
    int idx = blockIdx.x * 256 + threadIdx.x;
    int K = CIN * 9;
    if (idx >= COUT * K) return;
    int n = idx / K;
    int k = idx - n * K;
    int tap = k / CIN;
    int cin = k - tap * CIN;
    Wp[idx] = (half_t)w[((size_t)n * CIN + cin) * 9 + tap];
}

// Fold BN + biases into per-output-channel alpha/delta
__global__ __launch_bounds__(256) void prep_scale_kernel(
    const float* __restrict__ c1b, const float* __restrict__ gamma,
    const float* __restrict__ beta, const float* __restrict__ mean,
    const float* __restrict__ var, const float* __restrict__ c2b,
    float* __restrict__ alpha1, float* __restrict__ delta1,
    float* __restrict__ alpha2, float* __restrict__ delta2) {
    int t = threadIdx.x;
    if (t < EMB1C) {
        float sc = gamma[t] * rsqrtf(var[t] + 1e-5f);
        alpha1[t] = sc;
        delta1[t] = (c1b[t] - mean[t]) * sc + beta[t];
    }
    if (t < EMB2C) {
        alpha2[t] = 1.0f;
        delta2[t] = c2b[t];
    }
}

// ---------------------------------------------------------------------------
// x (NCHW f32) -> xt (pixel-major [p][256] f16), LDS-tiled 64x64 transpose.
// Both global reads and writes fully coalesced. Amortized 9x by im2col reuse.
// ---------------------------------------------------------------------------
__global__ __launch_bounds__(256) void transpose_x_kernel(const float* __restrict__ x,
                                                          half_t* __restrict__ xt) {
    __shared__ float tile[64][65];
    const int tid = threadIdx.x;
    const int p0 = blockIdx.x * 64;          // 9216 % 64 == 0 -> same batch
    const int c0 = blockIdx.y * 64;
    const int b = p0 / HWSZ;
    const int hw0 = p0 - b * HWSZ;
#pragma unroll
    for (int i = 0; i < 16; ++i) {
        int e = tid + i * 256;
        int cl = e >> 6, m = e & 63;
        tile[cl][m] = x[(size_t)(b * EMB1C + c0 + cl) * HWSZ + hw0 + m];
    }
    __syncthreads();
#pragma unroll
    for (int i = 0; i < 16; ++i) {
        int e = tid + i * 256;
        int m = e >> 6, cl = e & 63;
        xt[(size_t)(p0 + m) * EMB1C + c0 + cl] = (half_t)tile[cl][m];
    }
}

// ---------------------------------------------------------------------------
// Async staging of one A tile (128 pixels x 64 k) into LDS buffer `sel`.
// 4 x GLOBAL_LOAD_ASYNC_TO_LDS_B128 per thread (ASYNCcnt-tracked); border
// rows zero-filled with plain LDS stores. Single __shared__ array in the
// kernel -> its LDS offset is 0, so byte offsets are valid LDS addresses.
// ---------------------------------------------------------------------------
#define LSTRIDE 72                 // LDS row stride in halfs (144 B, 16B align)
#define BUFH (128 * LSTRIDE)       // halfs per buffer

__device__ __forceinline__ void stage_tile(const half_t* __restrict__ in_,
                                           half_t* Asg, int sel, int kbase,
                                           int sm, int sk0, int pb, int py, int px) {
    const int tap = kbase >> 8;              // CIN=256: chunk inside one tap
    const int cin_base = kbase & 255;
    const int dy = tap / 3 - 1;
    const int dx = tap - (tap / 3) * 3 - 1;
    const int yy = py + dy, xx = px + dx;
    const bool ok = (yy >= 0) && (yy < HH) && (xx >= 0) && (xx < WW);
    const half_t* src = in_ + (size_t)(pb * HWSZ + yy * WW + xx) * EMB1C + cin_base;
    const int base = sel * BUFH + sm * LSTRIDE;
#pragma unroll
    for (int i = 0; i < 4; ++i) {
        const int o = sk0 + 2 * i;           // 16-byte octet index 0..7
        if (ok) {
            unsigned lds = (unsigned)((base + o * 8) * 2);
            asm volatile("global_load_async_to_lds_b128 %0, %1, off"
                         :: "v"(lds), "v"(src + o * 8) : "memory");
        } else {
            v8h z = {};
            *(v8h*)&Asg[base + o * 8] = z;
        }
    }
}

// ---------------------------------------------------------------------------
// Implicit-GEMM 3x3 SAME conv via WMMA f32_16x16x32_f16 (CIN=256, K=2304).
//   Block tile: 128 pixels x 64 out-channels, K-chunk 64 (inside one tap).
//   8 waves, each owning a 32x32 output (2x2 subtiles) -> 8 WMMAs per stage.
//   Double-buffered async LDS staging: stage kt+1 issued before computing kt,
//   one s_wait_asynccnt + one barrier per stage (loads hide behind WMMAs).
//   Input: f16 pixel-major [p][256].  B frags straight from Wp[n][K] (in L2).
//   Output: f16 pixel-major [p][COUT], epilogue y = acc*alpha[c] + delta[c].
// ---------------------------------------------------------------------------
__global__ __launch_bounds__(256) void conv_gemm_kernel(
    const half_t* __restrict__ in_, const half_t* __restrict__ Wp,
    half_t* __restrict__ out, const float* __restrict__ alpha,
    const float* __restrict__ delta, int COUT) {
    constexpr int K = 9 * EMB1C;   // 2304
    __shared__ __attribute__((aligned(16))) half_t As[2 * BUFH];

    const int tid = threadIdx.x;
    const int p0 = blockIdx.x * 128;  // pixel tile base
    const int N0 = blockIdx.y * 64;   // out-channel tile base

    const int wv = tid >> 5, lane = tid & 31;
    const int r = lane & 15, hf = lane >> 4;
    const int wm = (wv >> 1) * 32;    // wave M offset: 0,32,64,96
    const int wn = (wv & 1) * 32;     // wave N offset: 0,32

    // staging: fixed pixel row per thread
    const int sm  = tid & 127;
    const int sk0 = tid >> 7;
    const int p  = p0 + sm;
    const int pb = p / HWSZ;
    const int phw = p - pb * HWSZ;
    const int py = phw / WW;
    const int px = phw - py * WW;

    const half_t* WpRow0 = Wp + (size_t)(N0 + wn + r) * K + hf * 16;
    const half_t* WpRow1 = Wp + (size_t)(N0 + wn + 16 + r) * K + hf * 16;

    v8f acc00 = {}, acc01 = {}, acc10 = {}, acc11 = {};

    // prologue: stage 0 into buffer 0
    stage_tile(in_, As, 0, 0, sm, sk0, pb, py, px);
    asm volatile("s_wait_asynccnt 0x0" ::: "memory");
    __syncthreads();

    constexpr int NST = K / 64;       // 36 stages
    for (int kt = 0; kt < NST; ++kt) {
        const int sel = kt & 1;
        if (kt + 1 < NST)
            stage_tile(in_, As, sel ^ 1, (kt + 1) * 64, sm, sk0, pb, py, px);

        const int kbase = kt * 64;
        const half_t* Abase = As + sel * BUFH;
#pragma unroll
        for (int step = 0; step < 2; ++step) {
            const int ks = step * 32;
            v16h a0, a1;
            {
                const half_t* ar = Abase + (wm + r) * LSTRIDE + ks + hf * 16;
                v8h lo = *(const v8h*)ar, hi = *(const v8h*)(ar + 8);
#pragma unroll
                for (int i = 0; i < 8; ++i) { a0[i] = lo[i]; a0[i + 8] = hi[i]; }
            }
            {
                const half_t* ar = Abase + (wm + 16 + r) * LSTRIDE + ks + hf * 16;
                v8h lo = *(const v8h*)ar, hi = *(const v8h*)(ar + 8);
#pragma unroll
                for (int i = 0; i < 8; ++i) { a1[i] = lo[i]; a1[i + 8] = hi[i]; }
            }
            v16h b0 = *(const v16h*)(WpRow0 + kbase + ks);
            v16h b1 = *(const v16h*)(WpRow1 + kbase + ks);
            acc00 = __builtin_amdgcn_wmma_f32_16x16x32_f16(
                false, a0, false, b0, (short)0, acc00, false, false);
            acc01 = __builtin_amdgcn_wmma_f32_16x16x32_f16(
                false, a0, false, b1, (short)0, acc01, false, false);
            acc10 = __builtin_amdgcn_wmma_f32_16x16x32_f16(
                false, a1, false, b0, (short)0, acc10, false, false);
            acc11 = __builtin_amdgcn_wmma_f32_16x16x32_f16(
                false, a1, false, b1, (short)0, acc11, false, false);
        }
        asm volatile("s_wait_asynccnt 0x0" ::: "memory");
        __syncthreads();
    }

    // ---- epilogue: D layout -> pixel-major f16, fused scale/shift ----
    const int col0 = N0 + wn + r;
    const int col1 = col0 + 16;
    const float al0 = alpha[col0], dl0 = delta[col0];
    const float al1 = alpha[col1], dl1 = delta[col1];
#pragma unroll
    for (int g = 0; g < 8; ++g) {
        const int row0 = p0 + wm + g + hf * 8;
        const int row1 = row0 + 16;
        out[(size_t)row0 * COUT + col0] = (half_t)(acc00[g] * al0 + dl0);
        out[(size_t)row0 * COUT + col1] = (half_t)(acc01[g] * al1 + dl1);
        out[(size_t)row1 * COUT + col0] = (half_t)(acc10[g] * al0 + dl0);
        out[(size_t)row1 * COUT + col1] = (half_t)(acc11[g] * al1 + dl1);
    }
}

// ---------------------------------------------------------------------------
// Per-pixel L2 norm over 128 channels (one wave per pixel)
// ---------------------------------------------------------------------------
__global__ __launch_bounds__(256) void l2norm_kernel(const half_t* __restrict__ src,
                                                     half_t* __restrict__ dst) {
    int wv = threadIdx.x >> 5, lane = threadIdx.x & 31;
    int p = blockIdx.x * 8 + wv;
    const half_t* sp = src + (size_t)p * EMB2C + lane * 4;
    float v0 = (float)sp[0], v1 = (float)sp[1], v2 = (float)sp[2], v3 = (float)sp[3];
    float s = v0 * v0 + v1 * v1 + v2 * v2 + v3 * v3;
#pragma unroll
    for (int off = 16; off > 0; off >>= 1) s += __shfl_xor(s, off, 32);
    float inv = 1.0f / fmaxf(sqrtf(s), 1e-12f);
    half_t* dp = dst + (size_t)p * EMB2C + lane * 4;
    dp[0] = (half_t)(v0 * inv); dp[1] = (half_t)(v1 * inv);
    dp[2] = (half_t)(v2 * inv); dp[3] = (half_t)(v3 * inv);
}

// ---------------------------------------------------------------------------
// Positive reps: normalize(rep_w + rep_b), write f32 copy + Ball column 4c
// ---------------------------------------------------------------------------
__global__ __launch_bounds__(128) void reps_kernel(const float* __restrict__ rep_w,
                                                   const float* __restrict__ rep_b,
                                                   float* __restrict__ reps_f32,
                                                   half_t* __restrict__ Ball) {
    int c = blockIdx.x, t = threadIdx.x;
    float v = rep_w[c * EMB2C + t] + rep_b[c * EMB2C + t];
    __shared__ float red[128];
    red[t] = v * v;
    __syncthreads();
    for (int s = 64; s > 0; s >>= 1) {
        if (t < s) red[t] += red[t + s];
        __syncthreads();
    }
    float inv = 1.0f / fmaxf(sqrtf(red[0]), 1e-12f);
    float nv = v * inv;
    reps_f32[c * EMB2C + t] = nv;
    Ball[(size_t)(c * 4) * EMB2C + t] = (half_t)nv;
}

// ---------------------------------------------------------------------------
// Neg-rep 3-layer MLP per (class, head), then normalize -> Ball col 4c+1+g
// ---------------------------------------------------------------------------
__global__ __launch_bounds__(128) void negmlp_kernel(const float* __restrict__ reps_f32,
                                                     const float* __restrict__ neg_w,
                                                     const float* __restrict__ neg_b,
                                                     half_t* __restrict__ Ball) {
    int c = blockIdx.x, g = blockIdx.y, j = threadIdx.x;
    __shared__ float hh[128];
    __shared__ float red[128];
    hh[j] = reps_f32[c * EMB2C + j];
    __syncthreads();
    for (int l = 0; l < 3; ++l) {
        const float* Wrow = neg_w + ((size_t)(g * 3 + l) * EMB2C + j) * EMB2C;
        float s = neg_b[(g * 3 + l) * EMB2C + j];
        for (int e = 0; e < EMB2C; ++e) s += Wrow[e] * hh[e];
        if (l < 2) s = fmaxf(s, 0.0f);
        __syncthreads();
        hh[j] = s;
        __syncthreads();
    }
    float v = hh[j];
    red[j] = v * v;
    __syncthreads();
    for (int s2 = 64; s2 > 0; s2 >>= 1) {
        if (j < s2) red[j] += red[j + s2];
        __syncthreads();
    }
    float inv = 1.0f / fmaxf(sqrtf(red[0]), 1e-12f);
    Ball[(size_t)(c * 4 + 1 + g) * EMB2C + j] = (half_t)(v * inv);
}

// ---------------------------------------------------------------------------
// dots[p][n] = embn[p][:] . Ball[n][:]   (73728 x 320 x 128)  pure WMMA
// wave = 16x16 tile; block covers 32 pixels x 64 cols; grid (2304, 5)
// ---------------------------------------------------------------------------
__global__ __launch_bounds__(256) void dot_gemm_kernel(const half_t* __restrict__ embn,
                                                       const half_t* __restrict__ Ball,
                                                       float* __restrict__ dots) {
    int tid = threadIdx.x;
    int wv = tid >> 5, lane = tid & 31;
    int r = lane & 15, hf = lane >> 4;
    int M0 = blockIdx.x * 32 + (wv >> 2) * 16;
    int N0 = blockIdx.y * 64 + (wv & 3) * 16;

    v8f acc = {};
#pragma unroll
    for (int kt = 0; kt < 4; ++kt) {
        const half_t* arow = embn + (size_t)(M0 + r) * EMB2C + kt * 32 + hf * 16;
        v8h alo = *(const v8h*)(arow);
        v8h ahi = *(const v8h*)(arow + 8);
        v16h a;
#pragma unroll
        for (int i = 0; i < 8; ++i) { a[i] = alo[i]; a[i + 8] = ahi[i]; }
        const half_t* brow = Ball + (size_t)(N0 + r) * EMB2C + kt * 32 + hf * 16;
        v16h bf = *(const v16h*)(brow);
        acc = __builtin_amdgcn_wmma_f32_16x16x32_f16(
            false, a, false, bf, (short)0, acc, false, false);
    }
    int col = N0 + r;
#pragma unroll
    for (int g = 0; g < 8; ++g)
        dots[(size_t)(M0 + g + hf * 8) * NCOLS + col] = acc[g];
}

// ---------------------------------------------------------------------------
// Head: per (b,c,hw) read 4 dots, produce the 5 outputs.
// Block stages 32 pixels x 320 dots through LDS (coalesced global reads).
// ---------------------------------------------------------------------------
__global__ __launch_bounds__(256) void head_kernel(const float* __restrict__ dots,
                                                   float* __restrict__ out) {
    __shared__ float tile[32 * 325];   // padded stride
    const int tid = threadIdx.x;
    const int p0 = blockIdx.x * 32;            // 9216 % 32 == 0 -> same batch
    const int b = p0 / HWSZ;
    const int hw0 = p0 - b * HWSZ;

    for (int e4 = tid; e4 < 32 * 80; e4 += 256) {
        int pr = e4 / 80;
        int co4 = e4 - pr * 80;
        const float4 v = *(const float4*)(dots + (size_t)(p0 + pr) * NCOLS + co4 * 4);
        float* d = &tile[pr * 325 + co4 * 4];
        d[0] = v.x; d[1] = v.y; d[2] = v.z; d[3] = v.w;
    }
    __syncthreads();

    const size_t TOT = (size_t)BB * NCLSC * HWSZ;   // 5,898,240
    for (int item = tid; item < 80 * 32; item += 256) {
        int c = item >> 5;
        int ml = item & 31;
        const float* dv = &tile[ml * 325 + c * 4];

        float d2 = fmaxf(2.0f - 2.0f * dv[0], 0.0f);
        float dist = sqrtf(d2);
        float minneg = 1e30f, maxpn = 0.0f;
        float dn[3];
#pragma unroll
        for (int m = 0; m < 3; ++m) {
            float dn2 = fmaxf(2.0f - 2.0f * dv[m + 1], 0.0f);
            dn[m] = sqrtf(dn2);
            minneg = fminf(minneg, dn[m]);
            maxpn = fmaxf(maxpn, expf(-dn2 * 2.0f));   // 1/(2*sigma^2)=2
        }
        int hw = hw0 + ml;
        size_t obase = (size_t)b * (NCLSC * HWSZ) + (size_t)c * HWSZ + hw;

        float shifted = dist + 0.3f * fmaxf(2.0f - minneg, 0.0f);
        float probs = expf(-shifted * shifted * 2.0f);
        float cs = fminf(fmaxf(probs, 0.0f), 1.0f);
        float cls = logf(fmaxf(cs, 1e-5f) / fmaxf(1.0f - cs, 1e-5f));

        out[obase] = cls;                        // cls_score
        out[TOT + obase] = maxpn;                // cls_score_neg
        out[2 * TOT + obase] = dist;             // distances
        size_t nbase = 3 * TOT + (size_t)b * (NCLSC * 3 * HWSZ)
                     + (size_t)c * (3 * HWSZ) + hw;
        out[nbase] = dn[0];                      // distances_neg
        out[nbase + HWSZ] = dn[1];
        out[nbase + 2 * HWSZ] = dn[2];
        out[6 * TOT + obase] = expf(-d2 * 2.0f); // probs_ori
    }
}

// ---------------------------------------------------------------------------
extern "C" void kernel_launch(void* const* d_in, const int* in_sizes, int n_in,
                              void* d_out, int out_size, void* d_ws, size_t ws_size,
                              hipStream_t stream) {
    const float* x     = (const float*)d_in[0];
    const float* w1    = (const float*)d_in[1];
    const float* b1    = (const float*)d_in[2];
    const float* gamma = (const float*)d_in[3];
    const float* beta  = (const float*)d_in[4];
    const float* mean  = (const float*)d_in[5];
    const float* var   = (const float*)d_in[6];
    const float* w2    = (const float*)d_in[7];
    const float* b2    = (const float*)d_in[8];
    const float* rep_w = (const float*)d_in[9];
    const float* rep_b = (const float*)d_in[10];
    const float* neg_w = (const float*)d_in[11];
    const float* neg_b = (const float*)d_in[12];
    float* out = (float*)d_out;

    char* ws = (char*)d_ws;
    size_t off = 0;
    auto alloc = [&](size_t bytes) -> void* {
        void* p = ws + off;
        off = (off + bytes + 255) & ~(size_t)255;
        return p;
    };
    half_t* Wp1      = (half_t*)alloc((size_t)2304 * EMB1C * 2);
    half_t* Wp2      = (half_t*)alloc((size_t)2304 * EMB2C * 2);
    float*  alpha1   = (float*)alloc(EMB1C * 4);
    float*  delta1   = (float*)alloc(EMB1C * 4);
    float*  alpha2   = (float*)alloc(EMB2C * 4);
    float*  delta2   = (float*)alloc(EMB2C * 4);
    half_t* xt       = (half_t*)alloc((size_t)NPIX * EMB1C * 2);
    half_t* hbuf     = (half_t*)alloc((size_t)NPIX * EMB1C * 2);
    half_t* embraw   = (half_t*)alloc((size_t)NPIX * EMB2C * 2);
    half_t* embn     = (half_t*)alloc((size_t)NPIX * EMB2C * 2);
    float*  reps_f32 = (float*)alloc((size_t)NCLSC * EMB2C * 4);
    half_t* Ball     = (half_t*)alloc((size_t)NCOLS * EMB2C * 2);
    float*  dots     = (float*)alloc((size_t)NPIX * NCOLS * 4);

    // weight packing + fused BN scales
    pack_w_kernel<<<(EMB1C * 2304 + 255) / 256, 256, 0, stream>>>(w1, Wp1, EMB1C, EMB1C);
    pack_w_kernel<<<(EMB2C * 2304 + 255) / 256, 256, 0, stream>>>(w2, Wp2, EMB2C, EMB1C);
    prep_scale_kernel<<<1, 256, 0, stream>>>(b1, gamma, beta, mean, var, b2,
                                             alpha1, delta1, alpha2, delta2);
    // class representatives
    reps_kernel<<<NCLSC, 128, 0, stream>>>(rep_w, rep_b, reps_f32, Ball);
    negmlp_kernel<<<dim3(NCLSC, 3), 128, 0, stream>>>(reps_f32, neg_w, neg_b, Ball);

    // x -> pixel-major f16 (feeds async-LDS conv staging)
    transpose_x_kernel<<<dim3(NPIX / 64, EMB1C / 64), 256, 0, stream>>>(x, xt);

    // conv1 (+BN fused) -> hbuf (f16, pixel-major)
    conv_gemm_kernel<<<dim3(NPIX / 128, EMB1C / 64), 256, 0, stream>>>(
        xt, Wp1, hbuf, alpha1, delta1, EMB1C);
    // conv2 (+bias) -> embraw (f16, pixel-major)
    conv_gemm_kernel<<<dim3(NPIX / 128, EMB2C / 64), 256, 0, stream>>>(
        hbuf, Wp2, embraw, alpha2, delta2, EMB2C);
    // per-pixel L2 norm
    l2norm_kernel<<<NPIX / 8, 256, 0, stream>>>(embraw, embn);
    // dots = embn x Ball^T
    dot_gemm_kernel<<<dim3(NPIX / 32, NCOLS / 64), 256, 0, stream>>>(embn, Ball, dots);
    // elementwise head -> 5 outputs
    head_kernel<<<NPIX / 32, 256, 0, stream>>>(dots, out);
}
